// _PtHeadSelection_73280732004918
// MI455X (gfx1250) — compile-verified, matrix operand order checked by reading, not run on hardware
//
#include <hip/hip_runtime.h>
#include <hip/hip_bf16.h>

// ---------------------------------------------------------------------------
// PtHeadSelection on MI455X (gfx1250, wave32, v_wmma_f32_16x16x32_f16).
//
// All matmul operands are either pre-swizzled into the CDNA5 WMMA operand
// layout (single 32B load) or read from row-major f16 via the two-contiguous-
// 8-element-runs property of the 16-bit A striping (2x b128 + shuffle).
// K5 stages its reused A tiles global->LDS with the Tensor Data Mover
// (tensor_load_to_lds + s_wait_tensorcnt), consumed via ds_load_b128.
// ---------------------------------------------------------------------------

typedef __attribute__((ext_vector_type(16))) _Float16 v16h;
typedef __attribute__((ext_vector_type(8)))  _Float16 v8h;
typedef __attribute__((ext_vector_type(8)))  float    v8f;

#define DEV __device__ __forceinline__
#define LN_THETA 9.210340371976184f  // ln(10000)

// WMMA 16-bit operand striping (ISA 7.12.2)
DEV int wmma_k(int pair, int hi) {
  return ((pair & 3) * 2) + (hi ? 8 : 0) + ((pair >> 2) * 16);
}
DEV int pk_hi(int k)   { return (k >> 3) & 1; }
DEV int pk_slot(int k) { return (k & 1) + 2 * (((k >> 4) << 2) + ((k & 7) >> 1)); }

DEV v8f wmma16(v16h a, v16h b, v8f c) {
  return __builtin_amdgcn_wmma_f32_16x16x32_f16(
      false, a, false, b, (short)0, c, false, false);
}

// Load a 16x32 operand block from row-major f16 (ld halfs per row):
// per lane two contiguous 16B runs (K = hi*8.. and 16+hi*8..) -> slots 0..15.
DEV v16h load_rm(const _Float16* p, int ld) {
  int l = threadIdx.x & 31, row = l & 15, hi = l >> 4;
  const _Float16* q = p + row * ld + hi * 8;
  v8h lo = *(const v8h*)q;
  v8h hv = *(const v8h*)(q + 16);
  return __builtin_shufflevector(lo, hv, 0, 1, 2, 3, 4, 5, 6, 7,
                                 8, 9, 10, 11, 12, 13, 14, 15);
}

// ---------------------------------------------------------------------------
// Tensor Data Mover: 2D f16 tile (rows x 256 halfs, contiguous rows) -> LDS.
// D# built per CDNA5 ISA: data_size=8B, dim0=64 units.
// This toolchain exposes the 6-arg builtin:
//   (uint32x4 g0, int32x8 g1, int32x4 g2, int32x4 g3, int32x8 g4, i32 cpol)
// ---------------------------------------------------------------------------
#if __has_builtin(__builtin_amdgcn_tensor_load_to_lds)
#define HAVE_TDM 1
typedef __attribute__((ext_vector_type(4))) unsigned tdm_u4;
typedef __attribute__((ext_vector_type(8))) int      tdm_i8;
typedef __attribute__((ext_vector_type(4))) int      tdm_i4;

DEV void tdm_load_tile(unsigned lds_byte_addr, const _Float16* gptr, int rows) {
  unsigned long long ga = (unsigned long long)(size_t)gptr;
  tdm_u4 g0;
  g0[0] = 1u;                                            // count=1 (valid D#)
  g0[1] = lds_byte_addr;                                 // lds_addr
  g0[2] = (unsigned)(ga & 0xffffffffu);                  // global_addr lo
  g0[3] = (unsigned)((ga >> 32) & 0x1ffffffu) | (2u << 30);  // addr hi | type=2
  tdm_i8 g1;
  g1[0] = (3 << 16);          // workgroup_mask=0 | data_size=3 (8B) | no flags
  g1[1] = (64 << 16);         // tensor_dim0 = 64 units (512B row)
  g1[2] = (rows << 16);       // tensor_dim0 hi=0 | tensor_dim1 lo = rows
  g1[3] = (64 << 16);         // tensor_dim1 hi=0 | tile_dim0 = 64
  g1[4] = rows;               // tile_dim1 = rows | tile_dim2 = 0
  g1[5] = 64;                 // tensor_dim0_stride = 64 units
  g1[6] = 0;
  g1[7] = 0;
  tdm_i4 z4 = {0, 0, 0, 0};   // groups 2/3 unused (2D tensor)
  tdm_i8 z8 = {0, 0, 0, 0, 0, 0, 0, 0};
  __builtin_amdgcn_tensor_load_to_lds(g0, g1, z4, z4, z8, 0);
}
#endif

// ---------------------------------------------------------------------------
// P0: pack a 256x256 f32 weight into WMMA-B f16 layout [dt(16)][kb(8)][32][16]
//     mode 0: B[k][d] = W[d*256+k]  (X @ W^T);  mode 1: B[k][d] = W[k*256+d]
// ---------------------------------------------------------------------------
__global__ void pack_weight_kernel(const float* __restrict__ W,
                                   _Float16* __restrict__ out, int mode) {
  int idx = blockIdx.x * 256 + threadIdx.x;   // 0..65535
  int slot = idx & 15, lane = (idx >> 4) & 31, kb = (idx >> 9) & 7, dt = idx >> 12;
  int k = kb * 32 + wmma_k(slot >> 1, lane >> 4) + (slot & 1);
  int d = dt * 16 + (lane & 15);
  out[idx] = (_Float16)(mode ? W[k * 256 + d] : W[d * 256 + k]);
}

// ---------------------------------------------------------------------------
// K1: qu = qz @ U^T, qv = qz @ V^T; RoPE; write packed operands:
//   quRpk : A-pack [n][h][stile][512]        (A of K2)
//   qvB2  : B-pack [n][h][ttile][512]        (B of K2)
//   qvB4  : B-pack [n][h][rt(2)][S/32][512]  (B of K4)
//   quOB4 : B-pack same shape                (B of K4)
// ---------------------------------------------------------------------------
__global__ void proj_rope_kernel(const float* __restrict__ qz,
                                 const _Float16* __restrict__ Upk,
                                 const _Float16* __restrict__ Vpk,
                                 _Float16* __restrict__ quRpk,
                                 _Float16* __restrict__ quOB4,
                                 _Float16* __restrict__ qvB2,
                                 _Float16* __restrict__ qvB4,
                                 int S, int innerN,
                                 long long sB, long long sI, long long sS) {
  __shared__ _Float16 As[8 * 32 * 16];   // 8 KB, operand-swizzled A tile
  int nsTiles = S >> 4;
  int n  = blockIdx.x / nsTiles;
  int s0 = (blockIdx.x % nsTiles) << 4;
  const float* arow = qz + (long long)(n / innerN) * sB +
                      (long long)(n % innerN) * sI + (long long)s0 * sS;

  for (int e = threadIdx.x; e < 2048; e += 256) {
    int row = e >> 7, k = (e & 127) * 2;
    float2 f = *(const float2*)(arow + (long long)row * sS + k);
    union { _Float16 h[2]; unsigned u; } pk;
    pk.h[0] = (_Float16)f.x; pk.h[1] = (_Float16)f.y;
    int kb = k >> 5, kk = k & 31;
    int lane = row + 16 * pk_hi(kk);
    *(unsigned*)&As[((kb * 32 + lane) << 4) + pk_slot(kk)] = pk.u;
  }
  __syncthreads();

  int wave = threadIdx.x >> 5;
  int l = threadIdx.x & 31, col = l & 15, hi = l >> 4;
  float invf = __expf(-(float)col * (LN_THETA / 16.0f));
  int skb = S >> 5;

  for (int job = wave; job < 16; job += 8) {   // (mat, head) jobs
    int mat = job >> 3, h = job & 7;
    const _Float16* Wp  = mat ? Vpk : Upk;
    const _Float16* BLo = Wp + ((2 * h)     * 8) * 512;
    const _Float16* BHi = Wp + ((2 * h + 1) * 8) * 512;
    v8f c_lo = {}, c_hi = {};
#pragma unroll 1
    for (int kb = 0; kb < 8; ++kb) {
      int o = (kb * 32 + l) << 4;
      v16h a = *(const v16h*)&As[o];
      c_lo = wmma16(a, *(const v16h*)(BLo + o), c_lo);
      c_hi = wmma16(a, *(const v16h*)(BHi + o), c_hi);
    }
    long long nh = (long long)n * 8 + h;
#pragma unroll
    for (int i = 0; i < 8; ++i) {
      int r = i + (hi ? 8 : 0);
      int s = s0 + r;
      float ang = (float)s * invf;
      float cs = __cosf(ang), sn = __sinf(ang);
      float xl = c_lo[i], xh = c_hi[i];
      float aLo = xl * cs - xh * sn, aHi = xh * cs + xl * sn;   // apply
      float oLo = xl * cs + xh * sn, oHi = xh * cs - xl * sn;   // apply_o
      int laneA = r + 16 * pk_hi(col);
      long long offA = ((nh * nsTiles + (s0 >> 4)) << 9) + (laneA << 4) + pk_slot(col);
      int kk = s & 31, kb2 = s >> 5;
      int laneB = col + 16 * pk_hi(kk);
      long long offB0 = (((nh * 2 + 0) * skb + kb2) << 9) + (laneB << 4) + pk_slot(kk);
      long long offB1 = (((nh * 2 + 1) * skb + kb2) << 9) + (laneB << 4) + pk_slot(kk);
      if (mat == 0) {
        quRpk[offA]     = (_Float16)aLo;
        quRpk[offA + 8] = (_Float16)aHi;
        quOB4[offB0]    = (_Float16)oLo;
        quOB4[offB1]    = (_Float16)oHi;
      } else {
        qvB2[offA]      = (_Float16)aLo;
        qvB2[offA + 8]  = (_Float16)aHi;
        qvB4[offB0]     = (_Float16)aLo;
        qvB4[offB1]     = (_Float16)aHi;
      }
    }
  }
}

// ---------------------------------------------------------------------------
// K2: F[n,h,s,t] = qu_rot . qv_rot (K=32, one WMMA) + mask  (fp32)
// ---------------------------------------------------------------------------
__global__ void f_logits_kernel(const _Float16* __restrict__ quRpk,
                                const _Float16* __restrict__ qvB2,
                                const float* __restrict__ mask,
                                float* __restrict__ F, int S) {
  int tiles = S >> 4;
  int bid = blockIdx.x;
  int t0 = (bid % tiles) << 4; bid /= tiles;
  int s0 = (bid % tiles) << 4; bid /= tiles;
  int h = bid & 7, n = bid >> 3;
  long long nh = (long long)n * 8 + h;
  int l = threadIdx.x & 31;

  v16h a = *(const v16h*)(quRpk + ((nh * tiles + (s0 >> 4)) << 9) + (l << 4));
  v16h b = *(const v16h*)(qvB2  + ((nh * tiles + (t0 >> 4)) << 9) + (l << 4));
  v8f c = {};
  c = wmma16(a, b, c);

  int col = l & 15, hi = l >> 4;
  const float* mrow = mask + (long long)n * S * S + (long long)s0 * S + t0;
  float* frow = F + ((nh * S + s0) * (long long)S) + t0;
#pragma unroll
  for (int i = 0; i < 8; ++i) {
    int r = i + (hi ? 8 : 0);
    frow[(long long)r * S + col] = c[i] + mrow[(long long)r * S + col];
  }
}

// ---------------------------------------------------------------------------
// K3: joint fp32 softmax over 160 candidates; writes packed-A qh layouts.
// ---------------------------------------------------------------------------
__global__ void softmax_kernel(const float* __restrict__ Ft,
                               const float* __restrict__ Fc,
                               _Float16* __restrict__ qhA_t,
                               _Float16* __restrict__ qhA2_t,
                               _Float16* __restrict__ qhA_c,
                               _Float16* __restrict__ qhA2_c) {
  const int P = 128, C = 32, H = 8;
  int row = blockIdx.x * (blockDim.x >> 5) + (threadIdx.x >> 5);
  int l = threadIdx.x & 31;
  int p = row & 127, c = (row >> 7) & 31, h = (row >> 12) & 7, b = row >> 15;
  long long nT = (long long)b * C + c;
  long long nC = (long long)b * P + p;
  long long baseT = ((nT * H + h) * P + p) * (long long)P;
  long long baseC = ((nC * H + h) * C + c) * (long long)C;

  float v[5];
#pragma unroll
  for (int j = 0; j < 4; ++j) v[j] = Ft[baseT + l + 32 * j] * 256.0f;
  v[4] = Fc[baseC + l] * 256.0f;

  float m = v[0];
#pragma unroll
  for (int j = 1; j < 5; ++j) m = fmaxf(m, v[j]);
#pragma unroll
  for (int off = 16; off; off >>= 1) m = fmaxf(m, __shfl_xor(m, off, 32));
  float sum = 0.0f;
#pragma unroll
  for (int j = 0; j < 5; ++j) { v[j] = __expf(v[j] - m); sum += v[j]; }
#pragma unroll
  for (int off = 16; off; off >>= 1) sum += __shfl_xor(sum, off, 32);
  float inv = 1.0f / sum;

  long long nhT = nT * 8 + h, nhC = nC * 8 + h;
#pragma unroll
  for (int j = 0; j < 4; ++j) {
    int t = l + 32 * j;
    _Float16 val = (_Float16)(v[j] * inv);
    long long iA = (((nhT * (P >> 4) + (p >> 4)) * (P >> 5) + (t >> 5)) << 9)
                 + (((p & 15) + 16 * pk_hi(t)) << 4) + pk_slot(t);
    qhA_t[iA] = val;
    long long iB = (((nhT * (P >> 4) + (t >> 4)) * (P >> 5) + (p >> 5)) << 9)
                 + (((t & 15) + 16 * pk_hi(p)) << 4) + pk_slot(p);
    qhA2_t[iB] = val;
  }
  {
    int t = l;
    _Float16 val = (_Float16)(v[4] * inv);
    long long iA = (((nhC * (C >> 4) + (c >> 4)) * (C >> 5) + (t >> 5)) << 9)
                 + (((c & 15) + 16 * pk_hi(t)) << 4) + pk_slot(t);
    qhA_c[iA] = val;
    long long iB = (((nhC * (C >> 4) + (t >> 4)) * (C >> 5) + (c >> 5)) << 9)
                 + (((t & 15) + 16 * pk_hi(c)) << 4) + pk_slot(c);
    qhA2_c[iB] = val;
  }
}

// ---------------------------------------------------------------------------
// K4: g1 = apply_o(qh @ qvR), g2 = apply(qh^T @ quO); packed operands.
// ---------------------------------------------------------------------------
__global__ void g_kernel(const _Float16* __restrict__ qhA,
                         const _Float16* __restrict__ qhA2,
                         const _Float16* __restrict__ qvB4,
                         const _Float16* __restrict__ quOB4,
                         _Float16* __restrict__ g1,
                         _Float16* __restrict__ g2, int S) {
  int tiles = S >> 4, skb = S >> 5;
  int bid = blockIdx.x;
  int s0 = (bid % tiles) << 4; bid /= tiles;
  int h = bid & 7, n = bid >> 3;
  long long nh = (long long)n * 8 + h;
  int l = threadIdx.x & 31, col = l & 15, hi = l >> 4;

  const _Float16* A1  = qhA   + (((nh * tiles + (s0 >> 4)) * skb) << 9);
  const _Float16* A2  = qhA2  + (((nh * tiles + (s0 >> 4)) * skb) << 9);
  const _Float16* B1l = qvB4  + (((nh * 2 + 0) * skb) << 9);
  const _Float16* B1h = qvB4  + (((nh * 2 + 1) * skb) << 9);
  const _Float16* B2l = quOB4 + (((nh * 2 + 0) * skb) << 9);
  const _Float16* B2h = quOB4 + (((nh * 2 + 1) * skb) << 9);

  v8f c1l = {}, c1h = {}, c2l = {}, c2h = {};
#pragma unroll 1
  for (int kb = 0; kb < skb; ++kb) {
    int o = (kb * 32 + l) << 4;
    if (kb + 1 < skb) {                       // -> global_prefetch_b8
      __builtin_prefetch(A1 + o + (32 << 4), 0, 0);
      __builtin_prefetch(B1l + o + (32 << 4), 0, 0);
    }
    v16h a1 = *(const v16h*)(A1 + o);
    v16h a2 = *(const v16h*)(A2 + o);
    c1l = wmma16(a1, *(const v16h*)(B1l + o), c1l);
    c1h = wmma16(a1, *(const v16h*)(B1h + o), c1h);
    c2l = wmma16(a2, *(const v16h*)(B2l + o), c2l);
    c2h = wmma16(a2, *(const v16h*)(B2h + o), c2h);
  }

  float invf = __expf(-(float)col * (LN_THETA / 16.0f));
  long long orow = (((long long)n * S + s0) << 8) + 32 * h + col;
#pragma unroll
  for (int i = 0; i < 8; ++i) {
    int r = i + (hi ? 8 : 0);
    float ang = (float)(s0 + r) * invf;
    float cs = __cosf(ang), sn = __sinf(ang);
    long long o = orow + ((long long)r << 8);
    float x1l = c1l[i], x1h = c1h[i], x2l = c2l[i], x2h = c2h[i];
    g1[o]      = (_Float16)(x1l * cs + x1h * sn);  // apply_o
    g1[o + 16] = (_Float16)(x1h * cs - x1l * sn);
    g2[o]      = (_Float16)(x2l * cs - x2h * sn);  // apply
    g2[o + 16] = (_Float16)(x2h * cs + x2l * sn);
  }
}

// ---------------------------------------------------------------------------
// K5: out = g1 @ U + g2 @ V (fp32). Row-major A tiles staged global->LDS via
//     the Tensor Data Mover (wave 0 issues 2 descriptors, s_wait_tensorcnt,
//     workgroup barrier); operands read as 2x ds_load_b128 per lane.
// ---------------------------------------------------------------------------
__global__ void out_kernel(const _Float16* __restrict__ g1,
                           const _Float16* __restrict__ g2,
                           const _Float16* __restrict__ Upk,   // N-mode pack
                           const _Float16* __restrict__ Vpk,
                           float* __restrict__ out,
                           int S, int innerN,
                           long long oB, long long oI, long long oS) {
  __shared__ _Float16 A1s[16 * 256];   // row-major tiles, 8 KB each
  __shared__ _Float16 A2s[16 * 256];
  int tiles = S >> 4;
  int n  = blockIdx.x / tiles;
  int s0 = (blockIdx.x % tiles) << 4;
  const _Float16* a1b = g1 + (((long long)n * S + s0) << 8);
  const _Float16* a2b = g2 + (((long long)n * S + s0) << 8);

#ifdef HAVE_TDM
  if (threadIdx.x < 32) {
    tdm_load_tile((unsigned)(size_t)&A1s[0], a1b, 16);
    tdm_load_tile((unsigned)(size_t)&A2s[0], a2b, 16);
    __builtin_amdgcn_s_wait_tensorcnt(0);
  }
#else
  for (int e = threadIdx.x; e < 1024; e += 256) {
    ((unsigned long long*)A1s)[e] = ((const unsigned long long*)a1b)[e];
    ((unsigned long long*)A2s)[e] = ((const unsigned long long*)a2b)[e];
  }
#endif
  __syncthreads();

  int wave = threadIdx.x >> 5;
  int l = threadIdx.x & 31, col = l & 15, hi = l >> 4;
  long long obase = (long long)(n / innerN) * oB +
                    (long long)(n % innerN) * oI + (long long)s0 * oS;

  for (int dt = wave; dt < 16; dt += 8) {
    const _Float16* Bu = Upk + (dt * 8) * 512;
    const _Float16* Bv = Vpk + (dt * 8) * 512;
    v8f c = {};
#pragma unroll 1
    for (int kb = 0; kb < 8; ++kb) {
      int o = (kb * 32 + l) << 4;
      if (kb + 1 < 8) __builtin_prefetch(Bu + o + (32 << 4), 0, 0);
      v16h a1 = load_rm(A1s + kb * 32, 256);
      v16h a2 = load_rm(A2s + kb * 32, 256);
      c = wmma16(a1, *(const v16h*)(Bu + o), c);
      c = wmma16(a2, *(const v16h*)(Bv + o), c);
    }
#pragma unroll
    for (int i = 0; i < 8; ++i) {
      int r = i + (hi ? 8 : 0);
      out[obase + (long long)r * oS + dt * 16 + col] = c[i];
    }
  }
}

// ---------------------------------------------------------------------------
extern "C" void kernel_launch(void* const* d_in, const int* in_sizes, int n_in,
                              void* d_out, int out_size, void* d_ws, size_t ws_size,
                              hipStream_t stream) {
  const float* qz     = (const float*)d_in[0];
  const float* mask_t = (const float*)d_in[1];
  const float* mask_c = (const float*)d_in[2];
  const float* u_t    = (const float*)d_in[3];
  const float* v_t    = (const float*)d_in[4];
  const float* u_c    = (const float*)d_in[5];
  const float* v_c    = (const float*)d_in[6];
  float* out = (float*)d_out;

  const long long E = 8LL * 1024 * 1024;
  char* w = (char*)d_ws;
  auto a16 = [&](long long n) { _Float16* p = (_Float16*)w; w += n * 2; return p; };
  auto a32 = [&](long long n) { float* p = (float*)w; w += n * 4; return p; };

  _Float16* quRpk_t = a16(E); _Float16* quOB4_t = a16(E);
  _Float16* qvB2_t  = a16(E); _Float16* qvB4_t  = a16(E);
  _Float16* quRpk_c = a16(E); _Float16* quOB4_c = a16(E);
  _Float16* qvB2_c  = a16(E); _Float16* qvB4_c  = a16(E);
  _Float16* g1_t = a16(E); _Float16* g2_t = a16(E);
  _Float16* g1_c = a16(E); _Float16* g2_c = a16(E);
  float* F_t = a32(4 * E);
  float* F_c = a32(E);
  _Float16* qhA_t  = a16(4 * E); _Float16* qhA2_t = a16(4 * E);
  _Float16* qhA_c  = a16(E);     _Float16* qhA2_c = a16(E);
  _Float16* upkT_t = a16(65536); _Float16* vpkT_t = a16(65536);
  _Float16* upkT_c = a16(65536); _Float16* vpkT_c = a16(65536);
  _Float16* upkN_t = a16(65536); _Float16* vpkN_t = a16(65536);
  _Float16* upkN_c = a16(65536); _Float16* vpkN_c = a16(65536);

  const int B = 8, C = 32, P = 128, D = 256;
  const long long PD = (long long)P * D, CPD = (long long)C * P * D;

  pack_weight_kernel<<<256, 256, 0, stream>>>(u_t, upkT_t, 0);
  pack_weight_kernel<<<256, 256, 0, stream>>>(v_t, vpkT_t, 0);
  pack_weight_kernel<<<256, 256, 0, stream>>>(u_c, upkT_c, 0);
  pack_weight_kernel<<<256, 256, 0, stream>>>(v_c, vpkT_c, 0);
  pack_weight_kernel<<<256, 256, 0, stream>>>(u_t, upkN_t, 1);
  pack_weight_kernel<<<256, 256, 0, stream>>>(v_t, vpkN_t, 1);
  pack_weight_kernel<<<256, 256, 0, stream>>>(u_c, upkN_c, 1);
  pack_weight_kernel<<<256, 256, 0, stream>>>(v_c, vpkN_c, 1);

  proj_rope_kernel<<<256 * (P / 16), 256, 0, stream>>>(
      qz, upkT_t, vpkT_t, quRpk_t, quOB4_t, qvB2_t, qvB4_t,
      P, 256, 0LL, PD, (long long)D);
  proj_rope_kernel<<<1024 * (C / 16), 256, 0, stream>>>(
      qz, upkT_c, vpkT_c, quRpk_c, quOB4_c, qvB2_c, qvB4_c,
      C, P, CPD, (long long)D, PD);

  f_logits_kernel<<<256 * 8 * (P / 16) * (P / 16), 32, 0, stream>>>(
      quRpk_t, qvB2_t, mask_t, F_t, P);
  f_logits_kernel<<<1024 * 8 * (C / 16) * (C / 16), 32, 0, stream>>>(
      quRpk_c, qvB2_c, mask_c, F_c, C);

  softmax_kernel<<<(B * 8 * C * P) / 8, 256, 0, stream>>>(
      F_t, F_c, qhA_t, qhA2_t, qhA_c, qhA2_c);

  g_kernel<<<256 * 8 * (P / 16), 32, 0, stream>>>(
      qhA_t, qhA2_t, qvB4_t, quOB4_t, g1_t, g2_t, P);
  g_kernel<<<1024 * 8 * (C / 16), 32, 0, stream>>>(
      qhA_c, qhA2_c, qvB4_c, quOB4_c, g1_c, g2_c, C);

  out_kernel<<<256 * (P / 16), 256, 0, stream>>>(
      g1_t, g2_t, upkN_t, vpkN_t, out, P, 256, 0LL, PD, (long long)D);
  out_kernel<<<1024 * (C / 16), 256, 0, stream>>>(
      g1_c, g2_c, upkN_c, vpkN_c, out + (long long)B * C * P * D,
      C, P, CPD, (long long)D, PD);
}
